// PredictModel_67267777790469
// MI455X (gfx1250) — matrix operand order repaired and checked by reading, not compiled, not run on hardware
//
#include <hip/hip_runtime.h>
#include <stdint.h>

#define NUM_CLASSES 21
#define TOP_K       200
#define PRE_NMS_K   256
#define NSORT       16384        // next pow2 >= 8732
#define NUM_PRIORS  8732
#define BATCH       32
#define CONF_THRESH 0.01f
#define NMS_THRESH  0.45f

typedef __attribute__((ext_vector_type(16))) _Float16 v16h;
typedef __attribute__((ext_vector_type(8)))  _Float16 v8h;
typedef __attribute__((ext_vector_type(8)))  float    v8f;

union V16 { v16h v; v8h h[2]; };

struct SM {
    union {
        unsigned long long keys[NSORT];          // 128 KB: sort phase
        _Float16 O[PRE_NMS_K * PRE_NMS_K];       // 128 KB: symmetric overlap-indicator (reuse)
    } u;
    float    ts  [PRE_NMS_K];                    // top scores
    float    bx1 [PRE_NMS_K];
    float    by1 [PRE_NMS_K];
    float    bx2 [PRE_NMS_K];
    float    by2 [PRE_NMS_K];
    float    area[PRE_NMS_K];
    _Float16 keepf[PRE_NMS_K];                   // keep vector as f16 0/1 (WMMA A operand)
    int      kflag[PRE_NMS_K];
    int      scan [PRE_NMS_K];
    int      keptidx[TOP_K];
};

__global__ __launch_bounds__(256)
void ssd_nms_kernel(const float* __restrict__ conf,
                    const float* __restrict__ loc,
                    const float* __restrict__ priors,
                    float* __restrict__ out)
{
    __shared__ SM sm;
    const int c   = blockIdx.x;      // class
    const int b   = blockIdx.y;      // image
    const int tid = threadIdx.x;

    float* outbc = out + ((size_t)b * NUM_CLASSES + c) * TOP_K * 5;

    if (c == 0) {                    // background class: all zeros
        for (int i = tid; i < TOP_K * 5; i += 256) outbc[i] = 0.0f;
        return;
    }

    // ---- 1) Async-stream this class's scores (stride-21 gather) into LDS key slots ----
    for (int p = tid; p < NUM_PRIORS; p += 256) {
        const float* gp = conf + ((size_t)b * NUM_PRIORS + p) * NUM_CLASSES + c;
        uint32_t ldsaddr = (uint32_t)(uintptr_t)(&sm.u.keys[p]);   // low 32b of flat shared addr = LDS offset
        asm volatile("global_load_async_to_lds_b32 %0, %1, off"
                     :: "v"(ldsaddr), "v"(gp) : "memory");
    }
    asm volatile("s_wait_asynccnt 0" ::: "memory");
    __syncthreads();

    // ---- 2) Pack sortable keys: (order-preserving score bits << 32) | (0xFFFFFFFF - idx) ----
    for (int p = tid; p < NSORT; p += 256) {
        if (p < NUM_PRIORS) {
            uint32_t bits = (uint32_t)(sm.u.keys[p] & 0xFFFFFFFFull);
            uint32_t m = (bits & 0x80000000u) ? ~bits : (bits | 0x80000000u);
            sm.u.keys[p] = ((unsigned long long)m << 32)
                         | (unsigned long long)(0xFFFFFFFFu - (uint32_t)p);
        } else {
            sm.u.keys[p] = 0ull;     // -inf padding
        }
    }
    __syncthreads();

    // ---- 3) In-LDS bitonic sort, descending (exact top_k incl. tie order) ----
    for (unsigned k = 2; k <= NSORT; k <<= 1) {
        for (unsigned j = k >> 1; j > 0; j >>= 1) {
            for (unsigned i = tid; i < NSORT; i += 256) {
                unsigned ij = i ^ j;
                if (ij > i) {
                    unsigned long long a = sm.u.keys[i], bb = sm.u.keys[ij];
                    bool desc = ((i & k) == 0);
                    bool sw   = desc ? (a < bb) : (a > bb);
                    if (sw) { sm.u.keys[i] = bb; sm.u.keys[ij] = a; }
                }
            }
            __syncthreads();
        }
    }

    // ---- 4) Extract top-256, decode only those boxes ----
    {
        unsigned long long key = sm.u.keys[tid];
        uint32_t um   = (uint32_t)(key >> 32);
        uint32_t bits = (um & 0x80000000u) ? (um & 0x7FFFFFFFu) : ~um;
        float    score = __uint_as_float(bits);
        uint32_t idx   = 0xFFFFFFFFu - (uint32_t)(key & 0xFFFFFFFFull);  // 8732>256 => always real

        float4 l4 = *(const float4*)(loc    + ((size_t)b * NUM_PRIORS + idx) * 4);
        float4 p4 = *(const float4*)(priors + (size_t)idx * 4);
        float cx = p4.x + l4.x * 0.1f * p4.z;
        float cy = p4.y + l4.y * 0.1f * p4.w;
        float w  = p4.z * __expf(l4.z * 0.2f);
        float h  = p4.w * __expf(l4.w * 0.2f);
        float x1 = cx - 0.5f * w, y1 = cy - 0.5f * h;
        float x2 = cx + 0.5f * w, y2 = cy + 0.5f * h;
        sm.ts[tid]  = score;
        sm.bx1[tid] = x1; sm.by1[tid] = y1;
        sm.bx2[tid] = x2; sm.by2[tid] = y2;
        sm.area[tid] = (x2 - x1) * (y2 - y1);
    }
    __syncthreads();

    // ---- 5) Symmetric overlap matrix O[j][i] = (IoU > 0.45) as f16 0/1 (overwrites keys) ----
    //         8 IoUs per iteration -> one aligned 16B LDS store (ds_store_b128)
    {
        const int j = tid;
        const float ax1 = sm.bx1[j], ay1 = sm.by1[j];
        const float ax2 = sm.bx2[j], ay2 = sm.by2[j];
        const float aa  = sm.area[j];
        for (int i0 = 0; i0 < PRE_NMS_K; i0 += 8) {
            v8h pack;
            #pragma unroll
            for (int e = 0; e < 8; ++e) {
                const int i = i0 + e;
                float x1 = fmaxf(ax1, sm.bx1[i]);
                float y1 = fmaxf(ay1, sm.by1[i]);
                float x2 = fminf(ax2, sm.bx2[i]);
                float y2 = fminf(ay2, sm.by2[i]);
                float inter = fmaxf(x2 - x1, 0.0f) * fmaxf(y2 - y1, 0.0f);
                float uni   = aa + sm.area[i] - inter;
                float iou   = inter / fmaxf(uni, 1e-12f);
                pack[e] = (_Float16)((iou > NMS_THRESH) ? 1.0f : 0.0f);
            }
            *(v8h*)&sm.u.O[j * PRE_NMS_K + i0] = pack;   // row start 512B-aligned
        }
        sm.keepf[tid] = (_Float16)0.0f;
        sm.kflag[tid] = 0;
    }
    __syncthreads();

    // ---- 6) Greedy NMS on wave 0; cross-block suppression via v_wmma_f32_16x16x32_f16 ----
    //         O is symmetric -> B fragment read row-wise as contiguous halves (ds_load_b128 x2)
    if (tid < 32) {
        const int lane = tid;
        const int n    = lane & 15;                 // candidate column within block
        const int aoff = (lane >= 16) ? 16 : 0;     // A-fragment K sub-offset
        const int koff = (lane >= 16) ? 16 : 0;     // B-fragment K sub-offset
        for (int t = 0; t < 16; ++t) {
            // sup[n] = sum_j keep[j] * O[j, 16t+n]  (keep is 0 for unfinalized j => exact)
            v8f acc = (v8f){0.f,0.f,0.f,0.f,0.f,0.f,0.f,0.f};
            #pragma unroll
            for (int uu = 0; uu < 8; ++uu) {
                const int j0 = uu * 32;
                // A 16x32 f16: element e -> K=j0+aoff+e  (rows broadcast)
                const _Float16* abase = &sm.keepf[j0 + aoff];
                V16 au;
                au.h[0] = *(const v8h*)(abase);
                au.h[1] = *(const v8h*)(abase + 8);
                // B 32x16 f16: element e -> O[(16t+n)][j0+koff+e]  (symmetry: == O[j0+koff+e][16t+n])
                const _Float16* bbase = &sm.u.O[(t * 16 + n) * PRE_NMS_K + j0 + koff];
                V16 bu;
                bu.h[0] = *(const v8h*)(bbase);
                bu.h[1] = *(const v8h*)(bbase + 8);
                acc = __builtin_amdgcn_wmma_f32_16x16x32_f16(
                          false, au.v, false, bu.v, (short)0, acc, false, false);
            }
            bool sup    = acc[0] > 0.5f;            // cross-block suppression
            bool mykeep = false;
            for (int m = 0; m < 16; ++m) {          // serial in-block tail (wave-uniform control)
                uint32_t bal  = __builtin_amdgcn_ballot_w32(sup);
                bool validm   = sm.ts[t * 16 + m] > CONF_THRESH;
                bool keep_m   = validm && (((bal >> m) & 1u) == 0u);
                if (keep_m)
                    sup = sup || ((float)sm.u.O[(t * 16 + m) * PRE_NMS_K + (t * 16 + n)] > 0.5f);
                if (n == m) mykeep = keep_m;
            }
            if (lane < 16) {
                sm.keepf[t * 16 + lane] = (_Float16)(mykeep ? 1.0f : 0.0f);
                sm.kflag[t * 16 + lane] = mykeep ? 1 : 0;
            }
        }
    }
    __syncthreads();

    // ---- 7) Stable compaction (kept-in-score-order first), write [200,5] rows ----
    sm.scan[tid] = sm.kflag[tid];
    __syncthreads();
    for (int off = 1; off < 256; off <<= 1) {
        int v = sm.scan[tid];
        if (tid >= off) v += sm.scan[tid - off];
        __syncthreads();
        sm.scan[tid] = v;
        __syncthreads();
    }
    const int total = sm.scan[PRE_NMS_K - 1];
    if (sm.kflag[tid]) {
        int pos = sm.scan[tid] - 1;
        if (pos < TOP_K) sm.keptidx[pos] = tid;
    }
    __syncthreads();

    if (tid < TOP_K) {
        float r0 = 0.f, r1 = 0.f, r2 = 0.f, r3 = 0.f, r4 = 0.f;
        if (tid < total) {
            int i = sm.keptidx[tid];
            r0 = sm.ts[i]; r1 = sm.bx1[i]; r2 = sm.by1[i]; r3 = sm.bx2[i]; r4 = sm.by2[i];
        }
        float* row = outbc + (size_t)tid * 5;
        row[0] = r0; row[1] = r1; row[2] = r2; row[3] = r3; row[4] = r4;
    }
}

extern "C" void kernel_launch(void* const* d_in, const int* in_sizes, int n_in,
                              void* d_out, int out_size, void* d_ws, size_t ws_size,
                              hipStream_t stream) {
    const float* conf   = (const float*)d_in[0];   // [32, 8732, 21]
    const float* loc    = (const float*)d_in[1];   // [32, 8732, 4]
    const float* priors = (const float*)d_in[2];   // [8732, 4]
    float* out = (float*)d_out;                    // [32, 21, 200, 5]
    (void)in_sizes; (void)n_in; (void)out_size; (void)d_ws; (void)ws_size;

    dim3 grid(NUM_CLASSES, BATCH);                 // one block per (class, image)
    dim3 block(256);
    ssd_nms_kernel<<<grid, block, 0, stream>>>(conf, loc, priors, out);
}
